// Generator_30803505446951
// MI455X (gfx1250) — matrix operand order repaired and checked by reading, not compiled, not run on hardware
//
#include <hip/hip_runtime.h>
#include <hip/hip_bf16.h>
#include <math.h>

// ---- problem constants (match reference) ----
#define TT   32
#define BB   64
#define DIN  512
#define HH   512
#define VV   32000
#define FF   768
#define G3   1536   // 3*H

typedef __attribute__((ext_vector_type(16))) __bf16 bf16x16;
typedef __attribute__((ext_vector_type(8)))  float  floatx8;

union FragU { uint4 u[2]; bf16x16 v; };

#define LDS_PAD 8   // bf16 elements of row padding: row stride K+8 -> bank rotation

// ---------------------------------------------------------------------------
// M=64 WMMA GEMM:  C[64,N] = act(A_bf16[64,K] @ W_bf16[N,K]^T + bias)
// Template NT = N-tiles (16 cols) per wave. Block = 256 threads (8 waves),
// workgroup covers N = 128*NT columns; grid.x = N/(128*NT).
// A staged in LDS once per workgroup (row-padded: stride K+8 rotates banks).
// Each wave owns a 64x(16*NT) strip: one A fragment (ds_load_b128 x2) feeds
// NT WMMAs, one B fragment (global_load_b128 x2) feeds 4 WMMAs.
// NT=2: 12 mem-ops per 8 v_wmma_f32_16x16x32_bf16.
// act: 0 = none, 1 = tanh. Optionally writes a bf16 shadow of C.
// Dynamic LDS: 64*(K+LDS_PAD)*2 bytes.
// ---------------------------------------------------------------------------
template <int NT>
__global__ __launch_bounds__(256)
void wmma_gemm64(const __bf16* __restrict__ A, const __bf16* __restrict__ W,
                 const float* __restrict__ bias, float* __restrict__ C,
                 __bf16* __restrict__ Cbf, int K, int N, int act)
{
    extern __shared__ char smem[];
    __bf16* Alds = (__bf16*)smem;              // [64][K + LDS_PAD]
    const int LK  = K + LDS_PAD;
    const int tid = threadIdx.x;

    // ---- stage A into LDS (16B chunks, 256 threads cooperatively) ----
    const int chunksPerRow = K >> 3;           // 8 bf16 per 16B chunk
    for (int c = tid; c < 64 * chunksPerRow; c += 256) {
        const int row = c / chunksPerRow;
        const int kc  = (c - row * chunksPerRow) << 3;
        uint4 d = *(const uint4*)(A + (size_t)row * K + kc);
        *(uint4*)(Alds + (size_t)row * LK + kc) = d;
    }
    __syncthreads();

    const int lane  = tid & 31;
    const int wave  = tid >> 5;                // 0..7 -> N strip
    const int ncol0 = (blockIdx.x * 8 + wave) * (16 * NT);
    const int l15   = lane & 15;
    const int lhi   = lane >> 4;

    // B fragment addressing (16-bit B 32x16 layout):
    //   lane holds column N=l15; lanes 0-15 K=[k0..k0+7],[k0+16..k0+23]; lanes 16-31 K=[k0+8..k0+15],[k0+24..k0+31]
    const __bf16* wptr[NT];
#pragma unroll
    for (int nt = 0; nt < NT; ++nt)
        wptr[nt] = W + (size_t)(ncol0 + nt * 16 + l15) * K + lhi * 8;

    // A fragment addressing (16-bit A 16x32 layout), read from LDS:
    //   lanes 0-15: row M, K-chunks [0..7],[16..23]; lanes 16-31: [8..15],[24..31]
    const __bf16* aBase = Alds + l15 * LK + lhi * 8;   // + mt*16*LK + k0

    floatx8 acc[4 * NT] = {};

    for (int k0 = 0; k0 < K; k0 += 32) {
        FragU fb[NT];
#pragma unroll
        for (int nt = 0; nt < NT; ++nt) {
            fb[nt].u[0] = *(const uint4*)(wptr[nt] + k0);
            fb[nt].u[1] = *(const uint4*)(wptr[nt] + k0 + 16);
            // prefetch B stream 4 K-steps (256B) ahead into near caches
            __builtin_prefetch(wptr[nt] + k0 + 128, 0, 3);
        }
#pragma unroll
        for (int mt = 0; mt < 4; ++mt) {
            FragU fa;
            const __bf16* ap = aBase + (size_t)(mt * 16) * LK + k0;
            fa.u[0] = *(const uint4*)(ap);
            fa.u[1] = *(const uint4*)(ap + 16);
#pragma unroll
            for (int nt = 0; nt < NT; ++nt) {
                acc[nt * 4 + mt] = __builtin_amdgcn_wmma_f32_16x16x32_bf16(
                        /*neg_a=*/false, fa.v, /*neg_b=*/false, fb[nt].v,
                        /*c_mod=*/(short)0, acc[nt * 4 + mt],
                        /*reuse_a=*/false, /*reuse_b=*/false);
            }
        }
    }

    // ---- epilogue ----
#pragma unroll
    for (int nt = 0; nt < NT; ++nt) {
        const int col  = ncol0 + nt * 16 + l15;
        const float bval = bias ? bias[col] : 0.0f;
#pragma unroll
        for (int mt = 0; mt < 4; ++mt) {
            const int rowbase = mt * 16 + lhi * 8;
#pragma unroll
            for (int r = 0; r < 8; ++r) {
                float v = acc[nt * 4 + mt][r] + bval;
                if (act == 1) v = tanhf(v);
                size_t idx = (size_t)(rowbase + r) * (size_t)N + (size_t)col;
                C[idx] = v;
                if (Cbf) Cbf[idx] = (__bf16)v;
            }
        }
    }
}

// ---------------------------------------------------------------------------
// fp32 -> bf16 conversion (weights, ctx)
// ---------------------------------------------------------------------------
__global__ void cvt_bf16(const float* __restrict__ in, __bf16* __restrict__ out, int n)
{
    int i = blockIdx.x * blockDim.x + threadIdx.x;
    if (i < n) out[i] = (__bf16)in[i];
}

// ---------------------------------------------------------------------------
// Embedding gather + bf16 convert: y[T*B] -> y_emb_bf[T*B,512]
// ---------------------------------------------------------------------------
__global__ void emb_gather(const int* __restrict__ y, const float* __restrict__ embW,
                           __bf16* __restrict__ out)
{
    int i = blockIdx.x * blockDim.x + threadIdx.x;   // < T*B*DIN
    int tb = i >> 9;
    int d  = i & 511;
    out[i] = (__bf16)embW[(size_t)y[tb] * DIN + d];
}

// ---------------------------------------------------------------------------
// Zero init of GRU state
// ---------------------------------------------------------------------------
__global__ void zero_state(float* __restrict__ h, __bf16* __restrict__ hbf)
{
    int i = blockIdx.x * blockDim.x + threadIdx.x;   // < B*H
    h[i] = 0.0f;
    hbf[i] = (__bf16)0.0f;
}

// ---------------------------------------------------------------------------
// GRU gate fusion: gi,gh [B,3H] -> h1 (state update, fp32 + bf16), x = h1*ct
// ---------------------------------------------------------------------------
__global__ void gru_fuse(const float* __restrict__ gi, const float* __restrict__ gh,
                         const float* __restrict__ ct, float* __restrict__ h,
                         __bf16* __restrict__ hbf, __bf16* __restrict__ xbf)
{
    int i = blockIdx.x * blockDim.x + threadIdx.x;   // < B*H
    int b = i >> 9;
    int j = i & 511;
    const float* gib = gi + (size_t)b * G3;
    const float* ghb = gh + (size_t)b * G3;
    float r = 1.0f / (1.0f + expf(-(gib[j] + ghb[j])));
    float z = 1.0f / (1.0f + expf(-(gib[HH + j] + ghb[HH + j])));
    float n = tanhf(gib[2 * HH + j] + r * ghb[2 * HH + j]);
    float h1 = (1.0f - z) * n + z * h[i];
    h[i]   = h1;
    hbf[i] = (__bf16)h1;
    xbf[i] = (__bf16)(h1 * ct[i]);
}

// ---------------------------------------------------------------------------
// 8-head self-attention over a single token (heads act as sequence).
// grid = B, block = 64. scores = q.k / 64, softmax over heads, weighted v.
// ---------------------------------------------------------------------------
__global__ void attn_core(const float* __restrict__ q, const float* __restrict__ k,
                          const float* __restrict__ v, float* __restrict__ att,
                          __bf16* __restrict__ attbf)
{
    __shared__ float sc[8][8];
    __shared__ float aw[8][8];
    const int b = blockIdx.x;
    const int t = threadIdx.x;              // 0..63
    const float* qb = q + (size_t)b * HH;
    const float* kb = k + (size_t)b * HH;
    const float* vb = v + (size_t)b * HH;
    const int hh = t >> 3, gg = t & 7;
    float s = 0.0f;
#pragma unroll 8
    for (int d = 0; d < 64; ++d) s += qb[hh * 64 + d] * kb[gg * 64 + d];
    sc[hh][gg] = s * (1.0f / 64.0f);
    __syncthreads();
    if (t < 8) {
        float m = sc[t][0];
        for (int g2 = 1; g2 < 8; ++g2) m = fmaxf(m, sc[t][g2]);
        float e[8], sum = 0.0f;
        for (int g2 = 0; g2 < 8; ++g2) { e[g2] = expf(sc[t][g2] - m); sum += e[g2]; }
        float inv = 1.0f / sum;
        for (int g2 = 0; g2 < 8; ++g2) aw[t][g2] = e[g2] * inv;
    }
    __syncthreads();
#pragma unroll
    for (int j = 0; j < 8; ++j) {
        int d = (t & 7) * 8 + j;
        float o = 0.0f;
#pragma unroll
        for (int g2 = 0; g2 < 8; ++g2) o += aw[hh][g2] * vb[g2 * 64 + d];
        size_t idx = (size_t)b * HH + hh * 64 + d;
        att[idx]   = o;
        attbf[idx] = (__bf16)o;
    }
}

// ---------------------------------------------------------------------------
// Row softmax over V=32000. grid = B, block = 256.
// ---------------------------------------------------------------------------
__global__ __launch_bounds__(256)
void softmax_vocab(const float* __restrict__ in, float* __restrict__ out)
{
    __shared__ float red[256];
    const int b = blockIdx.x;
    const float* row = in + (size_t)b * VV;
    float* orow = out + (size_t)b * VV;
    float m = -INFINITY;
    for (int i = threadIdx.x; i < VV; i += 256) m = fmaxf(m, row[i]);
    red[threadIdx.x] = m; __syncthreads();
    for (int s = 128; s > 0; s >>= 1) {
        if (threadIdx.x < s) red[threadIdx.x] = fmaxf(red[threadIdx.x], red[threadIdx.x + s]);
        __syncthreads();
    }
    m = red[0]; __syncthreads();
    float sum = 0.0f;
    for (int i = threadIdx.x; i < VV; i += 256) sum += expf(row[i] - m);
    red[threadIdx.x] = sum; __syncthreads();
    for (int s = 128; s > 0; s >>= 1) {
        if (threadIdx.x < s) red[threadIdx.x] += red[threadIdx.x + s];
        __syncthreads();
    }
    const float inv = 1.0f / red[0];
    for (int i = threadIdx.x; i < VV; i += 256) orow[i] = expf(row[i] - m) * inv;
}

// ---------------------------------------------------------------------------
// host-side orchestration
// ---------------------------------------------------------------------------
extern "C" void kernel_launch(void* const* d_in, const int* in_sizes, int n_in,
                              void* d_out, int out_size, void* d_ws, size_t ws_size,
                              hipStream_t stream)
{
    (void)in_sizes; (void)n_in; (void)out_size; (void)ws_size;

    const int*   y     = (const int*)  d_in[0];
    const float* ctx   = (const float*)d_in[1];
    const float* embW  = (const float*)d_in[2];
    const float* W_ih  = (const float*)d_in[3];
    const float* b_ih  = (const float*)d_in[4];
    const float* W_hh  = (const float*)d_in[5];
    const float* b_hh  = (const float*)d_in[6];
    const float* W_att = (const float*)d_in[7];
    const float* b_att = (const float*)d_in[8];
    const float* Wq    = (const float*)d_in[9];
    const float* bq    = (const float*)d_in[10];
    const float* Wk    = (const float*)d_in[11];
    const float* bk    = (const float*)d_in[12];
    const float* Wv    = (const float*)d_in[13];
    const float* bv    = (const float*)d_in[14];
    const float* Wfc   = (const float*)d_in[15];
    const float* bfc   = (const float*)d_in[16];
    const float* W_h2o = (const float*)d_in[17];
    const float* b_h2o = (const float*)d_in[18];
    const float* W_out = (const float*)d_in[19];
    const float* b_out = (const float*)d_in[20];
    float* out = (float*)d_out;

    // ---- workspace carve-up (256B aligned) ----
    char* ws = (char*)d_ws;
    size_t off = 0;
    auto alloc = [&](size_t bytes) -> void* {
        void* p = ws + off;
        off = (off + bytes + 255) & ~(size_t)255;
        return p;
    };
    __bf16* Wout_bf = (__bf16*)alloc((size_t)VV * DIN * 2);
    __bf16* Wih_bf  = (__bf16*)alloc((size_t)G3 * DIN * 2);
    __bf16* Whh_bf  = (__bf16*)alloc((size_t)G3 * HH * 2);
    __bf16* Watt_bf = (__bf16*)alloc((size_t)HH * FF * 2);
    __bf16* Wq_bf   = (__bf16*)alloc((size_t)HH * HH * 2);
    __bf16* Wk_bf   = (__bf16*)alloc((size_t)HH * HH * 2);
    __bf16* Wv_bf   = (__bf16*)alloc((size_t)HH * HH * 2);
    __bf16* Wfc_bf  = (__bf16*)alloc((size_t)HH * HH * 2);
    __bf16* Wh2o_bf = (__bf16*)alloc((size_t)DIN * HH * 2);
    __bf16* yemb_bf = (__bf16*)alloc((size_t)TT * BB * DIN * 2);
    __bf16* ctx_bf  = (__bf16*)alloc((size_t)BB * FF * 2);
    float*  ctv     = (float*) alloc((size_t)BB * HH * 4);
    float*  hst     = (float*) alloc((size_t)BB * HH * 4);
    __bf16* h_bf    = (__bf16*)alloc((size_t)BB * HH * 2);
    float*  gi      = (float*) alloc((size_t)BB * G3 * 4);
    float*  gh      = (float*) alloc((size_t)BB * G3 * 4);
    __bf16* x_bf    = (__bf16*)alloc((size_t)BB * HH * 2);
    float*  qf      = (float*) alloc((size_t)BB * HH * 4);
    float*  kf      = (float*) alloc((size_t)BB * HH * 4);
    float*  vf      = (float*) alloc((size_t)BB * HH * 4);
    float*  attf    = (float*) alloc((size_t)BB * HH * 4);
    __bf16* att_bf  = (__bf16*)alloc((size_t)BB * HH * 2);
    float*  h2f     = (float*) alloc((size_t)BB * HH * 4);
    __bf16* h2_bf   = (__bf16*)alloc((size_t)BB * HH * 2);
    float*  logitf  = (float*) alloc((size_t)BB * DIN * 4);
    __bf16* logit_bf= (__bf16*)alloc((size_t)BB * DIN * 2);
    float*  vscore  = (float*) alloc((size_t)BB * VV * 4);

    auto cvt = [&](const float* src, __bf16* dst, int n) {
        cvt_bf16<<<(n + 255) / 256, 256, 0, stream>>>(src, dst, n);
    };
    auto shbytes = [](int K) -> size_t { return (size_t)64 * (K + LDS_PAD) * 2; };

    // ---- one-time preprocessing (re-done every call; deterministic) ----
    cvt(W_out, Wout_bf, VV * DIN);
    cvt(W_ih,  Wih_bf,  G3 * DIN);
    cvt(W_hh,  Whh_bf,  G3 * HH);
    cvt(W_att, Watt_bf, HH * FF);
    cvt(Wq,    Wq_bf,   HH * HH);
    cvt(Wk,    Wk_bf,   HH * HH);
    cvt(Wv,    Wv_bf,   HH * HH);
    cvt(Wfc,   Wfc_bf,  HH * HH);
    cvt(W_h2o, Wh2o_bf, DIN * HH);
    cvt(ctx,   ctx_bf,  BB * FF);

    emb_gather<<<(TT * BB * DIN) / 256, 256, 0, stream>>>(y, embW, yemb_bf);
    zero_state<<<(BB * HH) / 256, 256, 0, stream>>>(hst, h_bf);

    // ct = tanh(ctx @ W_att^T + b_att): [64,768] x [512,768]^T  (N=512 -> NT=1)
    wmma_gemm64<1><<<HH / 128, 256, shbytes(FF), stream>>>(ctx_bf, Watt_bf, b_att, ctv,
                                                           nullptr, FF, HH, /*act=*/1);

    // ---- decode loop (stream-ordered; recurrence serializes) ----
    for (int t = 0; t < TT; ++t) {
        const __bf16* yt = yemb_bf + (size_t)t * BB * DIN;

        wmma_gemm64<2><<<G3 / 256, 256, shbytes(DIN), stream>>>(yt,   Wih_bf, b_ih, gi,
                                                                nullptr, DIN, G3, 0);
        wmma_gemm64<2><<<G3 / 256, 256, shbytes(HH), stream>>>(h_bf, Whh_bf, b_hh, gh,
                                                               nullptr, HH,  G3, 0);
        gru_fuse<<<(BB * HH) / 256, 256, 0, stream>>>(gi, gh, ctv, hst, h_bf, x_bf);

        wmma_gemm64<1><<<HH / 128, 256, shbytes(HH), stream>>>(x_bf, Wq_bf, bq, qf,
                                                               nullptr, HH, HH, 0);
        wmma_gemm64<1><<<HH / 128, 256, shbytes(HH), stream>>>(x_bf, Wk_bf, bk, kf,
                                                               nullptr, HH, HH, 0);
        wmma_gemm64<1><<<HH / 128, 256, shbytes(HH), stream>>>(x_bf, Wv_bf, bv, vf,
                                                               nullptr, HH, HH, 0);
        attn_core<<<BB, 64, 0, stream>>>(qf, kf, vf, attf, att_bf);

        wmma_gemm64<1><<<HH / 128, 256, shbytes(HH), stream>>>(att_bf, Wfc_bf, bfc, h2f,
                                                               h2_bf, HH, HH, 0);
        wmma_gemm64<1><<<DIN / 128, 256, shbytes(HH), stream>>>(h2_bf, Wh2o_bf, b_h2o,
                                                                logitf, logit_bf,
                                                                HH, DIN, /*act=*/1);
        // big one: [64,512] @ [32000,512]^T  (W_out bf16 stays L2-resident) NT=2
        wmma_gemm64<2><<<VV / 256, 256, shbytes(DIN), stream>>>(logit_bf, Wout_bf, b_out,
                                                                vscore, nullptr,
                                                                DIN, VV, 0);
        softmax_vocab<<<BB, 256, 0, stream>>>(vscore, out + (size_t)t * BB * VV);
    }
}
